// Complete3DRadiativeTransferSolver_27977416966536
// MI455X (gfx1250) — matrix operand order repaired and checked by reading, not compiled, not run on hardware
//
#include <hip/hip_runtime.h>
#include <cstdint>
#include <cstddef>

// Problem dimensions (fixed by the reference's setup_inputs()).
#define NBATCH 2
#define NZ 32
#define NY 48
#define NX 48
#define NWL 8
#define VOX (NZ * NY * NX)
#define NXP 50 /* x padded with one zero column on each side */

typedef __bf16 v16bf __attribute__((ext_vector_type(16)));
typedef float  v8f   __attribute__((ext_vector_type(8)));

__constant__ float c_lam_nm[NWL] = {8000.f, 9000.f, 10000.f, 11000.f,
                                    11500.f, 12000.f, 13000.f, 14000.f};

// float -> bf16 (round to nearest even), stored as raw u16.
__device__ __forceinline__ unsigned short f2bf(float f) {
  unsigned int u = __float_as_uint(f);
  u += 0x7FFFu + ((u >> 16) & 1u);
  return (unsigned short)(u >> 16);
}

// ---------------------------------------------------------------------------
// Zero-fill (activation pads must read as 0 for SAME conv).
// ---------------------------------------------------------------------------
__global__ void k_fill_zero(uint4* __restrict__ p, size_t n) {
  size_t i = (size_t)blockIdx.x * blockDim.x + threadIdx.x;
  size_t stride = (size_t)gridDim.x * blockDim.x;
  uint4 z; z.x = 0u; z.y = 0u; z.z = 0u; z.w = 0u;
  for (; i < n; i += stride) p[i] = z;
}

// ---------------------------------------------------------------------------
// Pack conv weights into the WMMA A-fragment layout (bf16).
// wp index = ((mf*27 + tap)*CBLKS + cb)*512 + lane*16 + h
// A 16x32 bf16 layout: lane group g = lane>>4, row m = lane&15,
//   half h -> K = 16*g + h (matches contiguous B fragment loads)
// ---------------------------------------------------------------------------
__global__ void k_repack(const float* __restrict__ w, unsigned short* __restrict__ wp,
                         int COUT, int CIN, int MFRAGS, int CBLKS) {
  int total = MFRAGS * 27 * CBLKS * 512;
  for (int idx = blockIdx.x * blockDim.x + threadIdx.x; idx < total;
       idx += gridDim.x * blockDim.x) {
    int h    = idx & 15;
    int lane = (idx >> 4) & 31;
    int rest = idx >> 9;
    int cb   = rest % CBLKS;
    int tap  = (rest / CBLKS) % 27;
    int mf   = rest / (CBLKS * 27);
    int g = lane >> 4, m = lane & 15;
    int cout = mf * 16 + m;
    int k  = 16 * g + h;
    int ci = cb * 32 + k;
    float v = 0.f;
    if (cout < COUT && ci < CIN)
      v = w[(size_t)(cout * CIN + ci) * 27 + tap];
    wp[idx] = f2bf(v);
  }
}

// ---------------------------------------------------------------------------
// Radiative transfer: Planck source + up/down short-characteristic sweeps.
// One thread per (b, wavelength, y, x) column; z fully unrolled in registers.
// Writes fp32 rt[b][w][z][y][x] and the bf16 conv1 input (channel = w).
// ---------------------------------------------------------------------------
__global__ void k_rt(const float* __restrict__ T, const float* __restrict__ sza,
                     float* __restrict__ rt, unsigned short* __restrict__ act0) {
  int idx = blockIdx.x * blockDim.x + threadIdx.x;
  if (idx >= NBATCH * NWL * NY * NX) return;
  int x = idx % NX;
  int y = (idx / NX) % NY;
  int w = (idx / (NX * NY)) % NWL;
  int b = idx / (NX * NY * NWL);

  const float Hp = 6.62607015e-34f, Cl = 2.99792458e8f, Kb = 1.380649e-23f;
  float lam  = c_lam_nm[w] * 1e-9f;
  float lam2 = lam * lam;
  float lam5 = lam2 * lam2 * lam;
  float c1 = 2.f * Hp * Cl * Cl / lam5;      // 2 h c^2 / lambda^5
  float cx = Hp * Cl / (lam * Kb);           // h c / (lambda k)

  // |mu| for the 4 up (== 4 down) streams; tau = ext * DZ = 1e-4 * 100
  float amu[4] = {0.125f, 0.375f, 0.625f, 0.875f};
  float tr[4];
#pragma unroll
  for (int a = 0; a < 4; ++a) tr[a] = expf(-0.01f / amu[a]);

  float Isurf = (0.3f / 3.14159265358979323846f) * fmaxf(cosf(sza[b]), 0.f);

  float sarr[NZ];
  float accz[NZ];
  float up[4];
#pragma unroll
  for (int a = 0; a < 4; ++a) up[a] = Isurf;

  const float* Tcol = T + (size_t)b * VOX + (size_t)y * NX + x;
#pragma unroll
  for (int z = 0; z < NZ; ++z) {               // surface -> TOA
    float Tt = Tcol[(size_t)z * (NY * NX)];
    float s  = c1 / expm1f(cx / Tt);           // Planck source
    sarr[z]  = s;
    float sum = 0.f;
#pragma unroll
    for (int a = 0; a < 4; ++a) { up[a] = up[a] * tr[a] + s * (1.f - tr[a]); sum += up[a]; }
    accz[z] = sum;
  }
  float dn[4] = {0.f, 0.f, 0.f, 0.f};
#pragma unroll
  for (int z = NZ - 1; z >= 0; --z) {          // TOA -> surface
    float s = sarr[z];
    float sum = 0.f;
#pragma unroll
    for (int a = 0; a < 4; ++a) { dn[a] = dn[a] * tr[a] + s * (1.f - tr[a]); sum += dn[a]; }
    float v = (accz[z] + sum) * 0.125f;        // angle mean over 8 streams
    rt[(((size_t)b * NWL + w) * NZ + z) * (NY * NX) + (size_t)y * NX + x] = v;
    // conv1 input, fragment-native layout [b][z][y][cblk=0][x+1][ch=w]
    act0[((((size_t)b * NZ + z) * NY + y) * 1) * (NXP * 32) + (size_t)(x + 1) * 32 + w] =
        f2bf(v);
  }
}

// ---------------------------------------------------------------------------
// Implicit-GEMM 3x3x3 SAME conv via v_wmma_f32_16x16x32_bf16.
// Block = one (b,z,y) row of 48 x. Wave wv owns MPW M-frags (16 channels each)
// and all 3 N-frags (x tiles of 16): MPW*3 accumulators, so each K-step is
// (MPW A-loads + 3 B-loads) feeding 3*MPW back-to-back WMMAs (register
// blocking on M doubles B reuse vs one-frag-per-wave).
// K loop: 27 taps x CBLKS channel blocks of 32. B frags load straight from the
// fragment-native activation layout; A frags from the prepacked weights. EXEC
// is uniform around every WMMA (tap-skip branch depends only on blockIdx).
// ---------------------------------------------------------------------------
template <int NWAVES, int MPW, int CBLKS, int COUT, bool FINAL>
__global__ void __launch_bounds__(NWAVES * 32)
k_conv_wmma(const unsigned short* __restrict__ actIn,
            const unsigned short* __restrict__ wpack,
            const float* __restrict__ bias,
            float* __restrict__ out,
            const float* __restrict__ rt) {
  int lane = threadIdx.x & 31;
  int wv   = threadIdx.x >> 5;
  int bid  = blockIdx.x;
  int y = bid % NY;
  int z = (bid / NY) % NZ;
  int b = bid / (NY * NZ);
  int g = lane >> 4, n = lane & 15;

  v8f acc[MPW][3];
#pragma unroll
  for (int i = 0; i < MPW; ++i)
#pragma unroll
    for (int j = 0; j < 3; ++j)
      acc[i][j] = {0.f, 0.f, 0.f, 0.f, 0.f, 0.f, 0.f, 0.f};

  const unsigned short* wl[MPW];
#pragma unroll
  for (int i = 0; i < MPW; ++i)
    wl[i] = wpack + ((size_t)(wv * MPW + i) * 27 * CBLKS) * 512 + (size_t)lane * 16;

  for (int tap = 0; tap < 27; ++tap) {
    int dz = tap / 9 - 1;
    int dy = (tap / 3) % 3 - 1;
    int dx = tap % 3 - 1;
    int zz = z + dz, yy = y + dy;
    if (zz < 0 || zz >= NZ || yy < 0 || yy >= NY) continue;  // uniform per block
    const unsigned short* ab =
        actIn + ((((size_t)b * NZ + zz) * NY + yy) * CBLKS) * (NXP * 32) +
        (size_t)(n + dx + 1) * 32 + (size_t)g * 16;
#pragma unroll
    for (int cb = 0; cb < CBLKS; ++cb) {
      v16bf A[MPW];
#pragma unroll
      for (int i = 0; i < MPW; ++i)
        A[i] = *(const v16bf*)(const void*)(wl[i] + (size_t)(tap * CBLKS + cb) * 512);
      const unsigned short* bp = ab + (size_t)cb * (NXP * 32);
      v16bf B0 = *(const v16bf*)(const void*)(bp);
      v16bf B1 = *(const v16bf*)(const void*)(bp + 16 * 32);
      v16bf B2 = *(const v16bf*)(const void*)(bp + 32 * 32);
#pragma unroll
      for (int i = 0; i < MPW; ++i) {
        acc[i][0] = __builtin_amdgcn_wmma_f32_16x16x32_bf16(false, A[i], false, B0, (short)0, acc[i][0], false, false);
        acc[i][1] = __builtin_amdgcn_wmma_f32_16x16x32_bf16(false, A[i], false, B1, (short)0, acc[i][1], false, false);
        acc[i][2] = __builtin_amdgcn_wmma_f32_16x16x32_bf16(false, A[i], false, B2, (short)0, acc[i][2], false, false);
      }
    }
  }

  // C/D layout: VGPR r -> row m = r + 8*(lane>>4), column n = lane&15.
#pragma unroll
  for (int i = 0; i < MPW; ++i) {
#pragma unroll
    for (int r = 0; r < 8; ++r) {
      int m = r + 8 * g;
      int cout = (wv * MPW + i) * 16 + m;
      if (cout >= COUT) continue;
      float bv = bias[cout];
      float v0 = acc[i][0][r] + bv, v1 = acc[i][1][r] + bv, v2 = acc[i][2][r] + bv;
      size_t base = ((((size_t)b * COUT + cout) * NZ + z) * NY + y) * (size_t)NX;
      if (FINAL) {  // conv4: out = rt + tanh(conv); COUT == NWL so rt shares indexing
        out[base + n]      = tanhf(v0) + rt[base + n];
        out[base + 16 + n] = tanhf(v1) + rt[base + 16 + n];
        out[base + 32 + n] = tanhf(v2) + rt[base + 32 + n];
      } else {
        out[base + n]      = v0;
        out[base + 16 + n] = v1;
        out[base + 32 + n] = v2;
      }
    }
  }
}

// ---------------------------------------------------------------------------
// GroupNorm statistics: one block per (batch, group); group region is a
// contiguous chPerGroup*VOX float chunk. Population mean/var (ddof=0).
// ---------------------------------------------------------------------------
__global__ void k_gn_stats(const float* __restrict__ yv, float* __restrict__ stats,
                           int chPerGroup) {
  __shared__ float sh[256];
  __shared__ float sh2[256];
  int tid = threadIdx.x;
  size_t nEl = (size_t)chPerGroup * VOX;
  const float* base = yv + (size_t)blockIdx.x * nEl;
  float s = 0.f, s2 = 0.f;
  for (size_t i = tid; i < nEl; i += blockDim.x) {
    float v = base[i];
    s += v; s2 += v * v;
  }
  sh[tid] = s; sh2[tid] = s2;
  __syncthreads();
  for (int off = 128; off > 0; off >>= 1) {
    if (tid < off) { sh[tid] += sh[tid + off]; sh2[tid] += sh2[tid + off]; }
    __syncthreads();
  }
  if (tid == 0) {
    float inv = 1.f / (float)nEl;
    float mean = sh[0] * inv;
    float var  = sh2[0] * inv - mean * mean;
    stats[blockIdx.x * 2]     = mean;
    stats[blockIdx.x * 2 + 1] = var;
  }
}

// ---------------------------------------------------------------------------
// GroupNorm apply + SiLU, writing bf16 into the fragment-native activation
// layout (B-fragment channel order is natural: position = c & 31).
// ---------------------------------------------------------------------------
__global__ void k_gn_apply(const float* __restrict__ yv, const float* __restrict__ stats,
                           const float* __restrict__ gamma, const float* __restrict__ beta,
                           unsigned short* __restrict__ act, int C, int groups, int cblks) {
  size_t total = (size_t)NBATCH * C * VOX;
  size_t i = (size_t)blockIdx.x * blockDim.x + threadIdx.x;
  if (i >= total) return;
  int vox = (int)(i % VOX);
  int c   = (int)((i / VOX) % C);
  int b   = (int)(i / ((size_t)VOX * C));
  int chPerGroup = C / groups;
  int gidx = b * groups + c / chPerGroup;
  float mean = stats[gidx * 2], var = stats[gidx * 2 + 1];
  float v  = yv[i];
  float xn = (v - mean) * rsqrtf(var + 1e-5f) * gamma[c] + beta[c];
  float sl = xn / (1.f + expf(-xn));  // SiLU
  int x = vox % NX;
  int y = (vox / NX) % NY;
  int z = vox / (NX * NY);
  int cb = c >> 5;
  act[((((size_t)b * NZ + z) * NY + y) * cblks + cb) * (size_t)(NXP * 32) +
      (size_t)(x + 1) * 32 + (c & 31)] = f2bf(sl);
}

// ---------------------------------------------------------------------------
// Host launcher.
// ---------------------------------------------------------------------------
extern "C" void kernel_launch(void* const* d_in, const int* in_sizes, int n_in,
                              void* d_out, int out_size, void* d_ws, size_t ws_size,
                              hipStream_t stream) {
  (void)in_sizes; (void)n_in; (void)out_size; (void)ws_size;
  const float* T   = (const float*)d_in[0];
  /* d_in[1] = pressure: unused by the reference */
  const float* sza = (const float*)d_in[2];
  const float* w1 = (const float*)d_in[3];  const float* b1  = (const float*)d_in[4];
  const float* g1 = (const float*)d_in[5];  const float* be1 = (const float*)d_in[6];
  const float* w2 = (const float*)d_in[7];  const float* b2  = (const float*)d_in[8];
  const float* g2 = (const float*)d_in[9];  const float* be2 = (const float*)d_in[10];
  const float* w3 = (const float*)d_in[11]; const float* b3  = (const float*)d_in[12];
  const float* g3 = (const float*)d_in[13]; const float* be3 = (const float*)d_in[14];
  const float* w4 = (const float*)d_in[15]; const float* b4  = (const float*)d_in[16];
  float* out = (float*)d_out;

  char* ws = (char*)d_ws;
  size_t off = 0;
  auto take = [&](size_t bytes) -> char* {
    char* p = ws + off;
    off = (off + bytes + 255) & ~(size_t)255;
    return p;
  };

  float* rt = (float*)take((size_t)NBATCH * NWL * VOX * 4);
  size_t actBytes0 = (size_t)NBATCH * NZ * NY * 1 * NXP * 32 * 2;
  size_t actBytes1 = (size_t)NBATCH * NZ * NY * 4 * NXP * 32 * 2;
  size_t actBytes2 = (size_t)NBATCH * NZ * NY * 8 * NXP * 32 * 2;
  size_t actBytes3 = actBytes1;
  unsigned short* act0 = (unsigned short*)take(actBytes0);
  unsigned short* act1 = (unsigned short*)take(actBytes1);
  unsigned short* act2 = (unsigned short*)take(actBytes2);
  unsigned short* act3 = (unsigned short*)take(actBytes3);
  float* Y = (float*)take((size_t)NBATCH * 256 * VOX * 4);  // reused: y1, y2, y3
  unsigned short* wp1 = (unsigned short*)take((size_t)8  * 27 * 1 * 512 * 2);
  unsigned short* wp2 = (unsigned short*)take((size_t)16 * 27 * 4 * 512 * 2);
  unsigned short* wp3 = (unsigned short*)take((size_t)8  * 27 * 8 * 512 * 2);
  unsigned short* wp4 = (unsigned short*)take((size_t)1  * 27 * 4 * 512 * 2);
  float* stats = (float*)take(64 * 4);

  // 1) zero all activation buffers (contiguous region) so pads read as 0
  size_t zb = (size_t)(((char*)act3 + actBytes3) - (char*)act0);
  k_fill_zero<<<2048, 256, 0, stream>>>((uint4*)act0, zb / 16);

  // 2) weight repack into A-fragment layout
  k_repack<<<432,  256, 0, stream>>>(w1, wp1, 128, 8,   8,  1);
  k_repack<<<3456, 256, 0, stream>>>(w2, wp2, 256, 128, 16, 4);
  k_repack<<<3456, 256, 0, stream>>>(w3, wp3, 128, 256, 8,  8);
  k_repack<<<216,  256, 0, stream>>>(w4, wp4, 8,   128, 1,  4);

  // 3) radiative transfer -> rt (fp32) + conv1 input (bf16)
  k_rt<<<(NBATCH * NWL * NY * NX + 255) / 256, 256, 0, stream>>>(T, sza, rt, act0);

  dim3 gconv(NBATCH * NZ * NY);

  // 4) layer 1: conv(8->128) -> GN(8 groups) -> SiLU       (4 waves x 2 M-frags)
  k_conv_wmma<4, 2, 1, 128, false><<<gconv, 128, 0, stream>>>(act0, wp1, b1, Y, nullptr);
  k_gn_stats<<<NBATCH * 8, 256, 0, stream>>>(Y, stats, 16);
  {
    size_t tot = (size_t)NBATCH * 128 * VOX;
    k_gn_apply<<<(unsigned)((tot + 255) / 256), 256, 0, stream>>>(Y, stats, g1, be1, act1, 128, 8, 4);
  }
  // 5) layer 2: conv(128->256) -> GN(16 groups) -> SiLU    (8 waves x 2 M-frags)
  k_conv_wmma<8, 2, 4, 256, false><<<gconv, 256, 0, stream>>>(act1, wp2, b2, Y, nullptr);
  k_gn_stats<<<NBATCH * 16, 256, 0, stream>>>(Y, stats, 16);
  {
    size_t tot = (size_t)NBATCH * 256 * VOX;
    k_gn_apply<<<(unsigned)((tot + 255) / 256), 256, 0, stream>>>(Y, stats, g2, be2, act2, 256, 16, 8);
  }
  // 6) layer 3: conv(256->128) -> GN(8 groups) -> SiLU     (4 waves x 2 M-frags)
  k_conv_wmma<4, 2, 8, 128, false><<<gconv, 128, 0, stream>>>(act2, wp3, b3, Y, nullptr);
  k_gn_stats<<<NBATCH * 8, 256, 0, stream>>>(Y, stats, 16);
  {
    size_t tot = (size_t)NBATCH * 128 * VOX;
    k_gn_apply<<<(unsigned)((tot + 255) / 256), 256, 0, stream>>>(Y, stats, g3, be3, act3, 128, 8, 4);
  }
  // 7) layer 4: conv(128->8) fused with tanh + residual, writes d_out
  k_conv_wmma<1, 1, 4, 8, true><<<gconv, 32, 0, stream>>>(act3, wp4, b4, out, rt);
}
